// nHFC_53257594471014
// MI455X (gfx1250) — compile-verified, matrix operand order checked by reading, not run on hardware
//
#include <hip/hip_runtime.h>
#include <stdint.h>

#define NN   20000
#define EE   320000
#define EDIM 10

typedef __attribute__((ext_vector_type(8)))  __bf16   v8bf;
typedef __attribute__((ext_vector_type(16))) __bf16   v16bf;
typedef __attribute__((ext_vector_type(8)))  float    v8f;
typedef __attribute__((ext_vector_type(4)))  uint32_t u32x4;

#if defined(__has_builtin)
#if __has_builtin(__builtin_amdgcn_global_load_async_to_lds_b64) && \
    __has_builtin(__builtin_amdgcn_global_load_async_to_lds_b128)
#define HAVE_ASYNC_LDS 1
#endif
#endif

#if defined(HAVE_ASYNC_LDS) && defined(__HIP_DEVICE_COMPILE__)
typedef __attribute__((ext_vector_type(2))) int i32x2;
typedef __attribute__((ext_vector_type(4))) int i32x4;
typedef __attribute__((address_space(1))) i32x2 g_i32x2;
typedef __attribute__((address_space(3))) i32x2 l_i32x2;
typedef __attribute__((address_space(1))) i32x4 g_i32x4;
typedef __attribute__((address_space(3))) i32x4 l_i32x4;
// Wait until at most N async-LDS ops remain outstanding (imm operands).
__device__ __forceinline__ void wait_async_le5() {
#if __has_builtin(__builtin_amdgcn_s_wait_asynccnt)
  __builtin_amdgcn_s_wait_asynccnt(5);
#else
  asm volatile("s_wait_asynccnt 0x5" ::: "memory");
#endif
}
__device__ __forceinline__ void wait_async_le0() {
#if __has_builtin(__builtin_amdgcn_s_wait_asynccnt)
  __builtin_amdgcn_s_wait_asynccnt(0);
#else
  asm volatile("s_wait_asynccnt 0x0" ::: "memory");
#endif
}
#endif

// fp32 -> bf16, round-to-nearest-even
__device__ __forceinline__ __bf16 f2bf(float f) {
  union { float f; uint32_t u; } v; v.f = f;
  uint32_t r = (v.u + 0x7FFFu + ((v.u >> 16) & 1u)) >> 16;
  union { uint16_t s; __bf16 b; } o; o.s = (uint16_t)r;
  return o.b;
}

__device__ __forceinline__ v16bf cat16(v8bf lo, v8bf hi) {
  return __builtin_shufflevector(lo, hi, 0, 1, 2, 3, 4, 5, 6, 7,
                                 8, 9, 10, 11, 12, 13, 14, 15);
}

__global__ void k_fill0(float* __restrict__ p, int n) {
  int i = blockIdx.x * blockDim.x + threadIdx.x;
  if (i < n) p[i] = 0.f;
}

// O[n][k] (dense, K padded to kpad with zeros) = bf16(X[n*lda + k])
__global__ void k_cvt_a(const float* __restrict__ X, int lda, int din, int kpad,
                        __bf16* __restrict__ O, int nrows) {
  int i = blockIdx.x * blockDim.x + threadIdx.x;
  if (i >= nrows * kpad) return;
  int n = i / kpad, k = i - n * kpad;
  O[i] = f2bf((k < din) ? X[(size_t)n * lda + k] : 0.f);
}

// O[c][k] = bf16(W[k][c])  (transposed, K padded to kpad with zeros)
__global__ void k_cvt_wt(const float* __restrict__ W, int din, int dout, int kpad,
                         __bf16* __restrict__ O) {
  int i = blockIdx.x * blockDim.x + threadIdx.x;
  if (i >= dout * kpad) return;
  int c = i / kpad, k = i - c * kpad;
  O[i] = f2bf((k < din) ? W[(size_t)k * dout + c] : 0.f);
}

// Y[NN x dout] = Abf[NN x kpad] @ Wt[dout x kpad]^T, bf16 WMMA, f32 accumulate.
// Requires NN % 16 == 0 (holds: 20000). 4 waves/block; wave w owns a 16x32
// output strip at (mt*16, ng*128 + w*32): 1 A frag x 2 B frags -> 2 WMMAs/step.
// Double-buffered async global->LDS staging (5 ops/wave/group; FIFO ASYNCcnt):
// issue group k+1, then s_wait_asynccnt 5 waits only on group k, overlapping
// DMA with the WMMA consume. Fragments are two aligned 16B LDS loads each, per
// the ISA 16-bit A layout: v<4: K = 2v+8*half ; v>=4: K = 16+2(v-4)+8*half.
__global__ void k_gemm_wmma(const __bf16* __restrict__ Abf, const __bf16* __restrict__ Wt,
                            float* __restrict__ Y, int kpad, int dout, int mtiles) {
  __shared__ __attribute__((aligned(16))) __bf16 As[2][16][32];         // buf, M x K
  __shared__ __attribute__((aligned(16))) __bf16 Bs[2][4][2][16][32];   // buf, wave, sub, N x K
  const int tid   = threadIdx.x;
  const int w     = tid >> 5;
  const int lane  = tid & 31;
  const int half  = lane >> 4;
  const int m     = lane & 15;
  const int mt    = blockIdx.x % mtiles;
  const int ng    = blockIdx.x / mtiles;
  const int mBase = mt * 16;
  const int nBase = ng * 128 + w * 32;

  // A staging: thread -> (row, 8B segment); 128 threads cover the 1KB tile
  const int ar_ = tid >> 3, aseg = (tid & 7) << 2;   // aseg in elements
  const __bf16* agp = Abf + (size_t)(mBase + ar_) * kpad + aseg;
  __bf16* alp[2] = { &As[0][ar_][aseg], &As[1][ar_][aseg] };

  // B staging: 4 x 16B per lane cover this wave's 2KB (2 sub x 16 cols x 64B)
  const __bf16* bgp[4];
  __bf16* blp[2][4];
#pragma unroll
  for (int j = 0; j < 4; ++j) {
    int o    = j * 512 + lane * 16;      // byte offset in wave region
    int cidx = o >> 6;                   // 0..31 column
    int sub  = cidx >> 4, c = cidx & 15;
    int ke   = (o & 63) >> 1;            // element offset within 64B row
    int gcol = nBase + sub * 16 + c;
    if (gcol >= dout) gcol = 0;          // clamp; results masked at writeback
    bgp[j]    = Wt + (size_t)gcol * kpad + ke;
    blp[0][j] = &Bs[0][w][sub][c][ke];
    blp[1][j] = &Bs[1][w][sub][c][ke];
  }

  auto stage = [&](int ks, int buf) {
    const int kk = ks << 5;
#if defined(HAVE_ASYNC_LDS) && defined(__HIP_DEVICE_COMPILE__)
    __builtin_amdgcn_global_load_async_to_lds_b64(
        (g_i32x2*)(const void*)(agp + kk), (l_i32x2*)(void*)alp[buf], 0, 0);
#pragma unroll
    for (int j = 0; j < 4; ++j)
      __builtin_amdgcn_global_load_async_to_lds_b128(
          (g_i32x4*)(const void*)(bgp[j] + kk), (l_i32x4*)(void*)blp[buf][j], 0, 0);
#else
    *(uint64_t*)alp[buf] = *(const uint64_t*)(agp + kk);
#pragma unroll
    for (int j = 0; j < 4; ++j)
      *(u32x4*)blp[buf][j] = *(const u32x4*)(bgp[j] + kk);
#endif
  };

  v8f acc0 = {}, acc1 = {};
  const int ksteps = kpad >> 5;
  stage(0, 0);                            // prologue
  for (int ks = 0; ks < ksteps; ++ks) {
    const int cur = ks & 1;
    if (ks + 1 < ksteps) {
      stage(ks + 1, cur ^ 1);             // overlap next tile's DMA with compute
#if defined(HAVE_ASYNC_LDS) && defined(__HIP_DEVICE_COMPILE__)
      wait_async_le5();                   // waits only on group ks
#endif
    } else {
#if defined(HAVE_ASYNC_LDS) && defined(__HIP_DEVICE_COMPILE__)
      wait_async_le0();
#endif
    }
    __syncthreads();
    const v8bf* ar  = (const v8bf*)&As[cur][m][0];
    const v8bf* b0r = (const v8bf*)&Bs[cur][w][0][m][0];
    const v8bf* b1r = (const v8bf*)&Bs[cur][w][1][m][0];
    v16bf av  = cat16(ar[half],  ar[2 + half]);
    v16bf bv0 = cat16(b0r[half], b0r[2 + half]);
    v16bf bv1 = cat16(b1r[half], b1r[2 + half]);
    acc0 = __builtin_amdgcn_wmma_f32_16x16x32_bf16(false, av, false, bv0,
                                                   (short)0, acc0, false, false);
    acc1 = __builtin_amdgcn_wmma_f32_16x16x32_bf16(false, av, false, bv1,
                                                   (short)0, acc1, false, false);
    __syncthreads();                      // protect buf before next-next issue
  }
  // C/D layout: VGPR j -> row = j + 8*half, col = lane&15 (rows always valid)
#pragma unroll
  for (int j = 0; j < 8; ++j) {
    int row = mBase + j + 8 * half;
    int c0 = nBase + m, c1 = nBase + 16 + m;
    if (c0 < dout) Y[(size_t)row * dout + c0] = acc0[j];
    if (c1 < dout) Y[(size_t)row * dout + c1] = acc1[j];
  }
}

// One wave32 per edge: m = relu(y[src] + edge_attr@We); atomicAdd into agg[dst]
__global__ void k_edge(const int64_t* __restrict__ ei, const float* __restrict__ ea,
                       const float* __restrict__ We, const float* __restrict__ Y,
                       float* __restrict__ Agg, int nedges, int dout) {
  const int lane = threadIdx.x & 31;
  const int wid  = blockIdx.x * (blockDim.x >> 5) + (threadIdx.x >> 5);
  if (wid >= nedges) return;
  const int64_t s = ei[wid];
  const int64_t d = ei[(size_t)nedges + wid];
  const float* eap = ea + (size_t)wid * EDIM;
  float eav[EDIM];
#pragma unroll
  for (int k = 0; k < EDIM; ++k) eav[k] = eap[k];
  const float* yrow = Y + (size_t)s * dout;
  float* arow = Agg + (size_t)d * dout;
  __builtin_prefetch(yrow + lane, 0, 3);   // global_prefetch on gathered row
  for (int c = lane; c < dout; c += 32) {
    float ew = 0.f;
#pragma unroll
    for (int k = 0; k < EDIM; ++k) ew = fmaf(eav[k], We[(size_t)k * dout + c], ew);
    float mv = fmaxf(yrow[c] + ew, 0.f);
    atomicAdd(arow + c, mv);
  }
}

__global__ void k_bias(float* __restrict__ Agg, const float* __restrict__ b,
                       int nrows, int dout) {
  int i = blockIdx.x * blockDim.x + threadIdx.x;
  if (i < nrows * dout) Agg[i] += b[i % dout];
}

// H[n, 0:width] = G[n*gstride + d] * DW[n*496 + dwoff + d]
__global__ void k_mul_slice(const float* __restrict__ G, int gstride,
                            const float* __restrict__ DW, int dwoff,
                            float* __restrict__ H, int width, int nrows) {
  int i = blockIdx.x * blockDim.x + threadIdx.x;
  if (i >= nrows * width) return;
  int n = i / width, d = i - n * width;
  H[(size_t)n * width + d] = G[(size_t)n * gstride + d] * DW[(size_t)n * 496 + dwoff + d];
}

extern "C" void kernel_launch(void* const* d_in, const int* in_sizes, int n_in,
                              void* d_out, int out_size, void* d_ws, size_t ws_size,
                              hipStream_t stream) {
  const float*   x  = (const float*)d_in[0];
  const int64_t* ei = (const int64_t*)d_in[1];
  const float*   ea = (const float*)d_in[2];
  const float *Wn_in  = (const float*)d_in[3],  *We_in  = (const float*)d_in[4],  *b_in  = (const float*)d_in[5];
  const float *Wn_dw  = (const float*)d_in[6],  *We_dw  = (const float*)d_in[7],  *b_dw  = (const float*)d_in[8];
  const float *Wn_out = (const float*)d_in[9],  *We_out = (const float*)d_in[10], *b_out = (const float*)d_in[11];
  const float *Wn_pw[4], *We_pw[4], *b_pw[4];
  for (int i = 0; i < 4; ++i) {
    Wn_pw[i] = (const float*)d_in[12 + 3 * i];
    We_pw[i] = (const float*)d_in[13 + 3 * i];
    b_pw[i]  = (const float*)d_in[14 + 3 * i];
  }

  // Workspace layout: per-node fp32 arrays (<=41MB each) stay L2-resident
  float*  ws    = (float*)d_ws;
  float*  fused = ws;                          // N x 512
  float*  dw    = fused + (size_t)NN * 512;    // N x 496
  float*  ybuf  = dw    + (size_t)NN * 496;    // N x 512 (gemm output scratch)
  float*  gbuf  = ybuf  + (size_t)NN * 512;    // N x 256 (gf output for pw layers)
  float*  hA    = gbuf  + (size_t)NN * 256;    // N x 256
  float*  hB    = hA    + (size_t)NN * 256;    // N x 256
  __bf16* abf   = (__bf16*)(hB + (size_t)NN * 256);   // N x 512 bf16 (K-padded A)
  __bf16* wtbf  = abf + (size_t)NN * 512;             // 512 x 512 bf16 (W^T)

  auto gf = [&](const float* X, int lda, int din, int dout,
                const float* Wn, const float* We, const float* b, float* out) {
    int kpad = (din + 31) & ~31;
    int totA = NN * kpad, totW = dout * kpad, tot = NN * dout;
    k_cvt_a<<<(totA + 255) / 256, 256, 0, stream>>>(X, lda, din, kpad, abf, NN);
    k_cvt_wt<<<(totW + 255) / 256, 256, 0, stream>>>(Wn, din, dout, kpad, wtbf);
    k_fill0<<<(tot + 255) / 256, 256, 0, stream>>>(out, tot);
    int mtiles  = (NN + 15) / 16;
    int ngroups = (dout + 127) / 128;
    k_gemm_wmma<<<mtiles * ngroups, 128, 0, stream>>>(abf, wtbf, ybuf, kpad, dout, mtiles);
    k_edge<<<(EE + 7) / 8, 256, 0, stream>>>(ei, ea, We, ybuf, out, EE, dout);
    k_bias<<<(tot + 255) / 256, 256, 0, stream>>>(out, b, NN, dout);
  };

  const int DIMS[5] = {16, 32, 64, 128, 256};
  const int offs[6] = {0, 16, 48, 112, 240, 496};

  // fused = gf(x) : 256 -> 512
  gf(x, 256, 256, 512, Wn_in, We_in, b_in, fused);
  // dw_abc = gf(fused[:,16:512]) : 496 -> 496  (strided view via lda)
  gf(fused + 16, 512, 496, 496, Wn_dw, We_dw, b_dw, dw);
  // h = pwa * dw_list[0]
  {
    int tot = NN * 16;
    k_mul_slice<<<(tot + 255) / 256, 256, 0, stream>>>(fused, 512, dw, 0, hA, 16, NN);
  }
  float* hin = hA; float* hout = hB;
  for (int i = 0; i < 4; ++i) {
    gf(hin, DIMS[i], DIMS[i], DIMS[i + 1], Wn_pw[i], We_pw[i], b_pw[i], gbuf);
    int w2 = DIMS[i + 1], tot = NN * w2;
    k_mul_slice<<<(tot + 255) / 256, 256, 0, stream>>>(gbuf, w2, dw, offs[i + 1], hout, w2, NN);
    float* t = hin; hin = hout; hout = t;
  }
  // out = gf(h) : 256 -> 256, directly into d_out
  gf(hin, 256, 256, 256, Wn_out, We_out, b_out, (float*)d_out);
}
